// _CrNonLocalBlock_78082505441671
// MI455X (gfx1250) — matrix operand order repaired and checked by reading, not compiled, not run on hardware
//
#include <hip/hip_runtime.h>
#include <hip/hip_bf16.h>
#include <math.h>

#define EPSV 1e-5f

typedef __attribute__((ext_vector_type(16))) __bf16 v16bf;
typedef __attribute__((ext_vector_type(8)))  float  v8f;

struct alignas(16) Q16 { unsigned u[4]; };
union F16x16 { v16bf v; Q16 q[2]; };

// ---- WMMA fragment helpers (CDNA5 wave32 layouts, CDNA5 ISA 7.12.2) --
// A (16x32 bf16, row-major [M, ldk]): lane L: m=L%16, h=L/16;
//   elems j<8 -> K=8h+j ; j>=8 -> K=16+8h+(j-8)  => two 16B loads at +8h and +8h+16
__device__ inline v16bf load_a_frag(const __bf16* base, int ldk) {
  int lane = threadIdx.x & 31;
  int m = lane & 15, h = lane >> 4;
  const Q16* p = (const Q16*)(base + (size_t)m * ldk + 8 * h);
  F16x16 f; f.q[0] = p[0]; f.q[1] = p[2];
  return f.v;
}
// B (32x16 bf16), Bt stored row-major [Ncol, ldk]: lane L: n=L%16, h=L/16;
//   elems j<8 -> K=8h+j ; j>=8 -> K=16+8h+(j-8)  => two 16B loads at +8h and +8h+16
__device__ inline v16bf load_b_frag(const __bf16* base, int ldk) {
  int lane = threadIdx.x & 31;
  int n = lane & 15, h = lane >> 4;
  const Q16* p = (const Q16*)(base + (size_t)n * ldk + 8 * h);
  F16x16 f; f.q[0] = p[0]; f.q[1] = p[2];
  return f.v;
}
__device__ inline v8f wmma_bf16(v16bf a, v16bf b, v8f c) {
  return __builtin_amdgcn_wmma_f32_16x16x32_bf16(false, a, false, b, (short)0, c,
                                                 false, false);
}

// ---------------- elementwise prep kernels ----------------
__global__ void k_cast(const float* __restrict__ s, __bf16* __restrict__ d, int n) {
  for (int i = blockIdx.x * blockDim.x + threadIdx.x; i < n; i += gridDim.x * blockDim.x)
    d[i] = (__bf16)s[i];
}

// x [B,C,N] f32 -> xT [B*N, C] bf16
__global__ void k_xT(const float* __restrict__ x, __bf16* __restrict__ xT,
                     int B, int C, int Npix) {
  int tot = B * C * Npix;
  for (int i = blockIdx.x * blockDim.x + threadIdx.x; i < tot; i += gridDim.x * blockDim.x) {
    int n = i % Npix, c = (i / Npix) % C, b = i / (Npix * C);
    xT[((size_t)(b * Npix + n)) * C + c] = (__bf16)x[i];
  }
}

// conv_w [O,C2,3,3] f32 -> wperm [(kh*3+kw)][O][C2] bf16
__global__ void k_convw(const float* __restrict__ w, __bf16* __restrict__ wp, int O, int C2) {
  int tot = 9 * O * C2;
  for (int i = blockIdx.x * blockDim.x + threadIdx.x; i < tot; i += gridDim.x * blockDim.x) {
    int c2 = i % C2, o = (i / C2) % O, k9 = i / (C2 * O);
    wp[i] = (__bf16)w[((size_t)(o * C2 + c2)) * 9 + k9];
  }
}

// ---------------- projection GEMM (NT, bf16 in, bf16 out) ----------------
// out[m, ncol] = sum_k A[m,k]*Wt[ncol,k] + bias[ncol]
// MODE 0: out bf16 [M, Nc] row-major.  MODE 1: out bf16 [B][Nc][Npix] (transposed).
template <int MODE>
__global__ __attribute__((amdgpu_waves_per_eu(1)))
void k_proj(const __bf16* __restrict__ A, const __bf16* __restrict__ Wt,
            const float* __restrict__ bias, __bf16* __restrict__ out,
            int M, int Nc, int K, int Npix) {
  int wave = (blockIdx.x * blockDim.x + threadIdx.x) >> 5;
  int ntiles = Nc >> 4;
  int total = (M >> 4) * ntiles;
  if (wave >= total) return;
  int mt = wave / ntiles, nt = wave % ntiles;
  int lane = threadIdx.x & 31;
  int nl = lane & 15, h = lane >> 4;

  const __bf16* Abase = A + (size_t)(mt * 16) * K;
  const __bf16* Bbase = Wt + (size_t)(nt * 16) * K;
  v8f acc = {};
  for (int k = 0; k < K; k += 32) {
    v16bf a = load_a_frag(Abase + k, K);
    v16bf b = load_b_frag(Bbase + k, K);
    acc = wmma_bf16(a, b, acc);
  }
  float bv = bias[nt * 16 + nl];
#pragma unroll
  for (int i = 0; i < 8; ++i) acc[i] += bv;

  if (MODE == 0) {
#pragma unroll
    for (int i = 0; i < 8; ++i) {
      int row = mt * 16 + i + 8 * h;
      out[(size_t)row * Nc + nt * 16 + nl] = (__bf16)acc[i];
    }
  } else {
#pragma unroll
    for (int i = 0; i < 8; ++i) {
      int row = mt * 16 + i + 8 * h;        // global pixel over B*Npix
      int b = row / Npix, n = row % Npix;
      int ci = nt * 16 + nl;
      out[(((size_t)b * Nc) + ci) * Npix + n] = (__bf16)acc[i];
    }
  }
}

// ---------------- flash attention ----------------
// th [B][N][Ci] (Q), phiT [B][N][Ci] (K rows), g [B][Ci][N] (V^T), y [B][N][Ci] bf16.
// One wave = 16 query rows; full VGPR budget (no occupancy pressure -> no spills).
__global__ __attribute__((amdgpu_waves_per_eu(1)))
void k_attn(const __bf16* __restrict__ th, const __bf16* __restrict__ phiT,
            const __bf16* __restrict__ g, __bf16* __restrict__ y,
            int B, int Npix, int Ci) {
  __shared__ __align__(16) float  sS[8][16][32];
  __shared__ __align__(16) __bf16 sP[8][16][32];
  __shared__ float sAlpha[8][16];
  __shared__ float sL[8][16];

  int wave = (blockIdx.x * blockDim.x + threadIdx.x) >> 5;
  int wl = (threadIdx.x >> 5) & 7;
  int tilesPerB = Npix >> 4;
  if (wave >= B * tilesPerB) return;
  int b = wave / tilesPerB, mt = wave % tilesPerB;
  int lane = threadIdx.x & 31;
  int nl = lane & 15, h = lane >> 4;

  const __bf16* thb  = th   + (size_t)b * Npix * Ci;
  const __bf16* phib = phiT + (size_t)b * Npix * Ci;
  const __bf16* gb   = g    + (size_t)b * Ci * Npix;
  __bf16*       yb   = y    + (size_t)b * Npix * Ci;

  v8f Y[8];
#pragma unroll
  for (int f = 0; f < 8; ++f) Y[f] = (v8f){};
  float m_run = -1e30f, l_run = 0.0f;

  const __bf16* Arow = thb + (size_t)(mt * 16) * Ci;

  for (int ct = 0; ct < (Npix >> 5); ++ct) {
    // ---- scores: S[16 x 32] over Ci ----
    v8f s0 = {}, s1 = {};
    const __bf16* Bk0 = phib + (size_t)(ct * 32) * Ci;
    const __bf16* Bk1 = Bk0 + (size_t)16 * Ci;
#pragma unroll
    for (int kk = 0; kk < 128; kk += 32) {
      v16bf a  = load_a_frag(Arow + kk, Ci);
      v16bf b0 = load_b_frag(Bk0 + kk, Ci);
      v16bf b1 = load_b_frag(Bk1 + kk, Ci);
      s0 = wmma_bf16(a, b0, s0);
      s1 = wmma_bf16(a, b1, s1);
    }
    // stage scores to LDS in row-major (wave-private; DS ops are in-order per wave)
#pragma unroll
    for (int i = 0; i < 8; ++i) {
      sS[wl][i + 8 * h][nl]      = s0[i];
      sS[wl][i + 8 * h][16 + nl] = s1[i];
    }
    // ---- online softmax (lanes 0..15, one row each) ----
    if (lane < 16) {
      float mx = m_run;
#pragma unroll 8
      for (int j = 0; j < 32; ++j) mx = fmaxf(mx, sS[wl][lane][j]);
      float alpha = __expf(m_run - mx);
      float sum = 0.0f;
#pragma unroll 8
      for (int j = 0; j < 32; ++j) {
        float pv = __expf(sS[wl][lane][j] - mx);
        sum += pv;
        sP[wl][lane][j] = (__bf16)pv;
      }
      l_run = l_run * alpha + sum;
      m_run = mx;
      sAlpha[wl][lane] = alpha;
    }
    // ---- rescale + PV:  Y[16 x 128] += P(16x32) * V(32x128) ----
    float ar[8];
#pragma unroll
    for (int i = 0; i < 8; ++i) ar[i] = sAlpha[wl][i + 8 * h];
    v16bf pa = load_a_frag((const __bf16*)&sP[wl][0][0], 32);
#pragma unroll
    for (int f = 0; f < 8; ++f) {
#pragma unroll
      for (int i = 0; i < 8; ++i) Y[f][i] *= ar[i];
      v16bf bv = load_b_frag(gb + (size_t)(f * 16) * Npix + ct * 32, Npix);
      Y[f] = wmma_bf16(pa, bv, Y[f]);
    }
  }
  // ---- finalize: divide by l, store y [N, Ci] bf16 ----
  if (lane < 16) sL[wl][lane] = 1.0f / l_run;
  float lr[8];
#pragma unroll
  for (int i = 0; i < 8; ++i) lr[i] = sL[wl][i + 8 * h];
#pragma unroll
  for (int f = 0; f < 8; ++f)
#pragma unroll
    for (int i = 0; i < 8; ++i) {
      int row = mt * 16 + i + 8 * h;
      yb[(size_t)row * Ci + f * 16 + nl] = (__bf16)(Y[f][i] * lr[i]);
    }
}

// ---------------- W projection + BN -> Wy f32 [B,C,N] ----------------
// Block = 8 waves sharing one output-channel tile: the 16x128 bf16 weight tile
// is staged into LDS once via gfx1250 async copy (ASYNCcnt), then all waves
// read B fragments from LDS.
__global__ __attribute__((amdgpu_waves_per_eu(1)))
void k_wgemm(const __bf16* __restrict__ yv, const __bf16* __restrict__ Wt,
             const float* __restrict__ wb, const float* __restrict__ bg,
             const float* __restrict__ bb, const float* __restrict__ bm,
             const float* __restrict__ bv, float* __restrict__ Wy,
             int Mtot, int C, int Ci, int Npix) {
  __shared__ __align__(16) __bf16 sW[16 * 128];

  int ntiles = C >> 4;                     // 16
  int nt = blockIdx.x % ntiles;
  int mtg = blockIdx.x / ntiles;           // group of 8 consecutive m-tiles
  int wl = threadIdx.x >> 5;
  int mt = mtg * 8 + wl;
  int lane = threadIdx.x & 31;
  int nl = lane & 15, h = lane >> 4;

  // ---- async stage of W o-tile (16 x Ci bf16 = 4KB): one B128 per thread ----
  {
    int t = threadIdx.x;                   // 256 threads x 16B = 4KB
    const __bf16* gsrc = Wt + (size_t)(nt * 16) * Ci + t * 8;
    unsigned ldsa = (unsigned)(size_t)&sW[t * 8];
    unsigned long long ga = (unsigned long long)(size_t)gsrc;
    asm volatile("global_load_async_to_lds_b128 %0, %1, off"
                 :: "v"(ldsa), "v"(ga) : "memory");
    asm volatile("s_wait_asynccnt 0x0" ::: "memory");
  }
  __syncthreads();

  if (mt < (Mtot >> 4)) {
    const __bf16* Abase = yv + (size_t)(mt * 16) * Ci;
    v8f acc = {};
#pragma unroll
    for (int k = 0; k < 128; k += 32) {
      v16bf a = load_a_frag(Abase + k, Ci);
      v16bf b = load_b_frag((const __bf16*)sW + k, Ci);
      acc = wmma_bf16(a, b, acc);
    }
    int o = nt * 16 + nl;
    float scale = bg[o] * rsqrtf(bv[o] + EPSV);
    float shift = (wb[o] - bm[o]) * scale + bb[o];
#pragma unroll
    for (int i = 0; i < 8; ++i) {
      int row = mt * 16 + i + 8 * h;
      int b = row / Npix, n = row % Npix;
      Wy[(((size_t)b * C) + o) * Npix + n] = acc[i] * scale + shift;
    }
  }
}

// ---------------- channel attention: stats over concat([Wy, x]) ----------------
__global__ void k_stats(const float* __restrict__ Wy, const float* __restrict__ x,
                        float* __restrict__ avg, float* __restrict__ mx,
                        int C, int Npix) {
  int C2 = 2 * C;
  int b = blockIdx.x / C2, c2 = blockIdx.x % C2;
  const float* src = (c2 < C) ? (Wy + (((size_t)b * C) + c2) * Npix)
                              : (x  + (((size_t)b * C) + (c2 - C)) * Npix);
  float s = 0.0f, m = -1e30f;
  for (int i = threadIdx.x; i < Npix; i += blockDim.x) {
    float v = src[i]; s += v; m = fmaxf(m, v);
  }
  __shared__ float ss[256], sm[256];
  ss[threadIdx.x] = s; sm[threadIdx.x] = m;
  __syncthreads();
  for (int st = 128; st > 0; st >>= 1) {
    if ((int)threadIdx.x < st) {
      ss[threadIdx.x] += ss[threadIdx.x + st];
      sm[threadIdx.x] = fmaxf(sm[threadIdx.x], sm[threadIdx.x + st]);
    }
    __syncthreads();
  }
  if (threadIdx.x == 0) { avg[blockIdx.x] = ss[0] / Npix; mx[blockIdx.x] = sm[0]; }
}

// gate[b][c2] = sigmoid(fc2(relu(fc1(avg))) + fc2(relu(fc1(mx))))
__global__ void k_gate(const float* __restrict__ avg, const float* __restrict__ mx,
                       const float* __restrict__ f1w, const float* __restrict__ f1b,
                       const float* __restrict__ f2w, const float* __restrict__ f2b,
                       float* __restrict__ gate, int C2, int Cr) {
  __shared__ float hsum[64];
  int b = blockIdx.x, t = threadIdx.x;
  if (t < Cr) {
    float sa = f1b[t], sm = f1b[t];
    for (int c = 0; c < C2; ++c) {
      sa += f1w[t * C2 + c] * avg[b * C2 + c];
      sm += f1w[t * C2 + c] * mx[b * C2 + c];
    }
    hsum[t] = fmaxf(sa, 0.0f) + fmaxf(sm, 0.0f);
  }
  __syncthreads();
  if (t < C2) {
    float s = 2.0f * f2b[t];
    for (int r = 0; r < Cr; ++r) s += f2w[t * Cr + r] * hsum[r];
    gate[b * C2 + t] = 1.0f / (1.0f + __expf(-s));
  }
}

// gated, zero-padded, channels-last bf16: ginT [B][66*66][C2]
__global__ void k_gatepad(const float* __restrict__ Wy, const float* __restrict__ x,
                          const float* __restrict__ gate, __bf16* __restrict__ ginT,
                          int B, int C, int Npix) {
  int C2 = 2 * C, Npad = 66 * 66;
  int tot = B * Npad * C2;
  for (int i = blockIdx.x * blockDim.x + threadIdx.x; i < tot; i += gridDim.x * blockDim.x) {
    int c2 = i % C2, pix = (i / C2) % Npad, b = i / (C2 * Npad);
    int hp = pix / 66, wp = pix % 66;
    float val = 0.0f;
    if (hp >= 1 && hp <= 64 && wp >= 1 && wp <= 64) {
      int n = (hp - 1) * 64 + (wp - 1);
      float v = (c2 < C) ? Wy[(((size_t)b * C) + c2) * Npix + n]
                         : x[(((size_t)b * C) + (c2 - C)) * Npix + n];
      val = v * gate[b * C2 + c2];
    }
    ginT[i] = (__bf16)val;
  }
}

// 3x3 conv (implicit GEMM, 9 shifted K=512 passes) + BN + ReLU -> z f32 [B,C,64,64]
__global__ __attribute__((amdgpu_waves_per_eu(1)))
void k_conv(const __bf16* __restrict__ ginT, const __bf16* __restrict__ wp,
            const float* __restrict__ cb, const float* __restrict__ bg,
            const float* __restrict__ bb, const float* __restrict__ bm,
            const float* __restrict__ bv, float* __restrict__ out,
            int B, int C, int C2) {
  int wave = (blockIdx.x * blockDim.x + threadIdx.x) >> 5;
  // per batch: 64 (h) * 4 (w tiles of 16) * 16 (o tiles) waves
  if (wave >= B * 64 * 4 * (C >> 4)) return;
  int r = wave & 4095;
  int b = wave >> 12;
  int hh = r >> 6, wt = (r >> 4) & 3, ot = r & 15;
  int lane = threadIdx.x & 31;
  int nl = lane & 15, h = lane >> 4;

  const __bf16* gin = ginT + (size_t)b * (66 * 66) * C2;
  v8f acc = {};
#pragma unroll
  for (int kh = 0; kh < 3; ++kh) {
#pragma unroll
    for (int kw = 0; kw < 3; ++kw) {
      const __bf16* abase = gin + (size_t)((hh + kh) * 66 + wt * 16 + kw) * C2;
      const __bf16* bbase = wp + (size_t)((kh * 3 + kw) * C + ot * 16) * C2;
#pragma unroll 4
      for (int kc = 0; kc < C2; kc += 32) {
        __builtin_prefetch((const void*)(abase + kc + 256), 0, 1);
        v16bf a  = load_a_frag(abase + kc, C2);
        v16bf bf = load_b_frag(bbase + kc, C2);
        acc = wmma_bf16(a, bf, acc);
      }
    }
  }
  int o = ot * 16 + nl;
  float scale = bg[o] * rsqrtf(bv[o] + EPSV);
  float shift = (cb[o] - bm[o]) * scale + bb[o];
#pragma unroll
  for (int i = 0; i < 8; ++i) {
    int pix = hh * 64 + wt * 16 + i + 8 * h;
    out[(((size_t)b * C) + o) * 4096 + pix] = fmaxf(acc[i] * scale + shift, 0.0f);
  }
}

// =====================================================================
extern "C" void kernel_launch(void* const* d_in, const int* in_sizes, int n_in,
                              void* d_out, int out_size, void* d_ws, size_t ws_size,
                              hipStream_t stream) {
  (void)in_sizes; (void)n_in; (void)out_size; (void)ws_size;
  const int B = 2, C = 256, N = 4096, Ci = 128, C2 = 512, Cr = 32;
  const int Npad = 66 * 66;

  // inputs in setup_inputs() insertion order; params dict in insertion order
  const float* x1 = (const float*)d_in[0];
  const float* x2 = (const float*)d_in[1];
  const float* pw[6];  const float* pb[6];   // g1,g2,theta1,theta2,phi1,phi2
  for (int i = 0; i < 6; ++i) { pw[i] = (const float*)d_in[2 + 2 * i]; pb[i] = (const float*)d_in[3 + 2 * i]; }
  const float* W1p[6]; const float* W2p[6];  // w, b, bn_g, bn_b, bn_m, bn_v
  for (int i = 0; i < 6; ++i) { W1p[i] = (const float*)d_in[14 + i]; W2p[i] = (const float*)d_in[20 + i]; }
  const float* fu1[10]; const float* fu2[10]; // fc1_w,fc1_b,fc2_w,fc2_b,conv_w,conv_b,bn_g,bn_b,bn_m,bn_v
  for (int i = 0; i < 10; ++i) { fu1[i] = (const float*)d_in[26 + i]; fu2[i] = (const float*)d_in[36 + i]; }

  // ---- workspace partition (bf16/f32, 256B aligned) ----
  size_t off = 0;
  char* ws = (char*)d_ws;
  auto alloc = [&](size_t bytes) -> void* {
    void* p = ws + off; off += (bytes + 255) & ~(size_t)255; return p;
  };
  __bf16* xT1 = (__bf16*)alloc((size_t)B * N * C * 2);
  __bf16* xT2 = (__bf16*)alloc((size_t)B * N * C * 2);
  __bf16* wproj[6];
  for (int i = 0; i < 6; ++i) wproj[i] = (__bf16*)alloc((size_t)Ci * C * 2);
  __bf16* wW1 = (__bf16*)alloc((size_t)C * Ci * 2);
  __bf16* wW2 = (__bf16*)alloc((size_t)C * Ci * 2);
  __bf16* wcv1 = (__bf16*)alloc((size_t)9 * C * C2 * 2);
  __bf16* wcv2 = (__bf16*)alloc((size_t)9 * C * C2 * 2);
  __bf16* th1 = (__bf16*)alloc((size_t)B * N * Ci * 2);
  __bf16* th2 = (__bf16*)alloc((size_t)B * N * Ci * 2);
  __bf16* ph1 = (__bf16*)alloc((size_t)B * N * Ci * 2);
  __bf16* ph2 = (__bf16*)alloc((size_t)B * N * Ci * 2);
  __bf16* g1  = (__bf16*)alloc((size_t)B * Ci * N * 2);
  __bf16* g2  = (__bf16*)alloc((size_t)B * Ci * N * 2);
  __bf16* y1  = (__bf16*)alloc((size_t)B * N * Ci * 2);
  __bf16* y2  = (__bf16*)alloc((size_t)B * N * Ci * 2);
  float* Wy1 = (float*)alloc((size_t)B * C * N * 4);
  float* Wy2 = (float*)alloc((size_t)B * C * N * 4);
  float* avg1 = (float*)alloc((size_t)B * C2 * 4);
  float* mx1  = (float*)alloc((size_t)B * C2 * 4);
  float* gt1  = (float*)alloc((size_t)B * C2 * 4);
  float* avg2 = (float*)alloc((size_t)B * C2 * 4);
  float* mx2  = (float*)alloc((size_t)B * C2 * 4);
  float* gt2  = (float*)alloc((size_t)B * C2 * 4);
  __bf16* gin1 = (__bf16*)alloc((size_t)B * Npad * C2 * 2);
  __bf16* gin2 = (__bf16*)alloc((size_t)B * Npad * C2 * 2);

  // ---- 1. casts / transposes ----
  k_xT<<<4096, 256, 0, stream>>>(x1, xT1, B, C, N);
  k_xT<<<4096, 256, 0, stream>>>(x2, xT2, B, C, N);
  for (int i = 0; i < 6; ++i)
    k_cast<<<128, 256, 0, stream>>>(pw[i], wproj[i], Ci * C);
  k_cast<<<128, 256, 0, stream>>>(W1p[0], wW1, C * Ci);
  k_cast<<<128, 256, 0, stream>>>(W2p[0], wW2, C * Ci);
  k_convw<<<4608, 256, 0, stream>>>(fu1[4], wcv1, C, C2);
  k_convw<<<4608, 256, 0, stream>>>(fu2[4], wcv2, C, C2);

  // ---- 2. projections (M = B*N = 8192, Nc = 128, K = 256) ----
  {
    const int blocks = (8192 / 16) * (128 / 16) * 32 / 256;  // 512
    k_proj<0><<<blocks, 256, 0, stream>>>(xT1, wproj[2], pb[2], th1, B * N, Ci, C, N); // theta1
    k_proj<0><<<blocks, 256, 0, stream>>>(xT2, wproj[3], pb[3], th2, B * N, Ci, C, N); // theta2
    k_proj<0><<<blocks, 256, 0, stream>>>(xT1, wproj[4], pb[4], ph1, B * N, Ci, C, N); // phi1 (as [N,Ci])
    k_proj<0><<<blocks, 256, 0, stream>>>(xT2, wproj[5], pb[5], ph2, B * N, Ci, C, N); // phi2
    k_proj<1><<<blocks, 256, 0, stream>>>(xT1, wproj[0], pb[0], g1, B * N, Ci, C, N);  // g1 -> [Ci,N]
    k_proj<1><<<blocks, 256, 0, stream>>>(xT2, wproj[1], pb[1], g2, B * N, Ci, C, N);  // g2 -> [Ci,N]
  }

  // ---- 3. cross flash attention ----
  {
    const int blocks = (B * (N / 16)) * 32 / 256;  // 64
    k_attn<<<blocks, 256, 0, stream>>>(th1, ph2, g2, y1, B, N, Ci);
    k_attn<<<blocks, 256, 0, stream>>>(th2, ph1, g1, y2, B, N, Ci);
  }

  // ---- 4. W projection + BN (LDS-staged weight tile via async copy) ----
  {
    const int blocks = ((B * N / 16) / 8) * (C / 16);  // 64 * 16 = 1024
    k_wgemm<<<blocks, 256, 0, stream>>>(y1, wW1, W1p[1], W1p[2], W1p[3], W1p[4], W1p[5],
                                        Wy1, B * N, C, Ci, N);
    k_wgemm<<<blocks, 256, 0, stream>>>(y2, wW2, W2p[1], W2p[2], W2p[3], W2p[4], W2p[5],
                                        Wy2, B * N, C, Ci, N);
  }

  // ---- 5. channel attention + gated padded input ----
  k_stats<<<B * C2, 256, 0, stream>>>(Wy1, x1, avg1, mx1, C, N);
  k_stats<<<B * C2, 256, 0, stream>>>(Wy2, x2, avg2, mx2, C, N);
  k_gate<<<B, 512, 0, stream>>>(avg1, mx1, fu1[0], fu1[1], fu1[2], fu1[3], gt1, C2, Cr);
  k_gate<<<B, 512, 0, stream>>>(avg2, mx2, fu2[0], fu2[1], fu2[2], fu2[3], gt2, C2, Cr);
  k_gatepad<<<8192, 256, 0, stream>>>(Wy1, x1, gt1, gin1, B, C, N);
  k_gatepad<<<8192, 256, 0, stream>>>(Wy2, x2, gt2, gin2, B, C, N);

  // ---- 6. 3x3 conv + BN + ReLU -> outputs ----
  float* z1 = (float*)d_out;
  float* z2 = z1 + (size_t)B * C * N;
  {
    const int blocks = (B * 64 * 4 * (C / 16)) * 32 / 256;  // 1024
    k_conv<<<blocks, 256, 0, stream>>>(gin1, wcv1, fu1[5], fu1[6], fu1[7], fu1[8], fu1[9],
                                       z1, B, C, C2);
    k_conv<<<blocks, 256, 0, stream>>>(gin2, wcv2, fu2[5], fu2[6], fu2[7], fu2[8], fu2[9],
                                       z2, B, C, C2);
  }
}
